// LSTM_352187318652
// MI455X (gfx1250) — matrix operand order repaired and checked by reading, not compiled, not run on hardware
//
#include <hip/hip_runtime.h>

#define SEQ   1000
#define BATCH 256
#define HID   256
#define NL    10
#define GATES (4*HID)   // 1024
#define KTOT  (2*HID)   // 512 (concat of x-part and h-part)

typedef __attribute__((ext_vector_type(16))) _Float16 v16h;
typedef __attribute__((ext_vector_type(8)))  _Float16 v8h;
typedef __attribute__((ext_vector_type(8)))  float    v8f;

// A-fragment (16x32 f16): per lane, halves 0..7 = K kbase..+7, halves 8..15 = K kbase+16..+23
__device__ __forceinline__ v16h frag_a_h(const _Float16* __restrict__ row, int kbase) {
    v8h lo = *(const v8h*)(row + kbase);
    v8h hi = *(const v8h*)(row + kbase + 16);
    v16h r;
#pragma unroll
    for (int i = 0; i < 8; ++i) { r[i] = lo[i]; r[i + 8] = hi[i]; }
    return r;
}

// Same, converting fp32 source on the fly (layer 0 reads raw inputs, no staging pass)
__device__ __forceinline__ v16h frag_a_f(const float* __restrict__ row, int kbase) {
    v16h r;
#pragma unroll
    for (int i = 0; i < 8; ++i) {
        r[i]     = (_Float16)row[kbase + i];
        r[i + 8] = (_Float16)row[kbase + 16 + i];
    }
    return r;
}

__device__ __forceinline__ float sigm(float x) { return 1.0f / (1.0f + __expf(-x)); }

// Pack weights: Wc[l][k][g] (f16, K-major so B-fragments are contiguous 32B rows)
__global__ void pack_w(const float* __restrict__ Wih, const float* __restrict__ Whh,
                       _Float16* __restrict__ Wc) {
    size_t idx = (size_t)blockIdx.x * blockDim.x + threadIdx.x;
    const size_t N = (size_t)NL * KTOT * GATES;
    if (idx >= N) return;
    int    g  = (int)(idx % GATES);
    size_t t2 = idx / GATES;
    int    k  = (int)(t2 % KTOT);
    int    l  = (int)(t2 / KTOT);
    float v = (k < HID) ? Wih[((size_t)l * GATES + g) * HID + k]
                        : Whh[((size_t)l * GATES + g) * HID + (k - HID)];
    Wc[idx] = (_Float16)v;
}

__global__ void pack_b(const float* __restrict__ bih, const float* __restrict__ bhh,
                       float* __restrict__ bc) {
    int idx = blockIdx.x * blockDim.x + threadIdx.x;
    if (idx < NL * GATES) bc[idx] = bih[idx] + bhh[idx];
}

// One timestep; loops over 10 layers with a device-wide barrier between layers.
// Grid: 64 blocks (8 row-tiles x 8 hcol-tiles), 128 threads (4 waves).
// Wave w computes gate chunk w (i/f/g/o) for the block's 32x32 hidden tile.
__global__ void __launch_bounds__(128)
lstm_step(const float* __restrict__ xall, int t,
          const _Float16* __restrict__ Wc, const float* __restrict__ bc,
          const _Float16* __restrict__ scR, _Float16* __restrict__ scW,
          _Float16* __restrict__ cur0, _Float16* __restrict__ cur1,
          float* __restrict__ sh32, unsigned* __restrict__ ctr)
{
    __shared__ float smem[32][132];   // 32 rows x 128 gate cols, padded stride

    const int tid   = threadIdx.x;
    const int lane  = tid & 31;
    const int wv    = tid >> 5;        // gate chunk 0..3
    const int lrow  = lane & 15;
    const int lhalf = lane >> 4;
    const int m0 = (blockIdx.x >> 3) * 32;   // batch rows
    const int h0 = (blockIdx.x & 7) * 32;    // hidden cols

    const float* xt = xall + (size_t)t * (BATCH * HID);

    for (int j = 0; j < NL; ++j) {
        const _Float16* curR = (j & 1) ? cur1 : cur0;      // written by previous layer
        _Float16*       curW = (j & 1) ? cur0 : cur1;      // read by next layer
        const _Float16* scRj = scR + (size_t)j * (BATCH * HID);
        const _Float16* Wl   = Wc + (size_t)j * KTOT * GATES;
        const int g0 = wv * HID + h0;
        const int kb = lhalf * 8;

        v8f acc00 = {}, acc01 = {}, acc10 = {}, acc11 = {};

        // ---- x-part of K (k = 0..255): branch-free straight-line loops ----
        if (j == 0) {
            const float* ar0 = xt + (size_t)(m0 + lrow) * HID;
            const float* ar1 = xt + (size_t)(m0 + 16 + lrow) * HID;
#pragma unroll 4
            for (int kk = 0; kk < 8; ++kk) {
                const int k0 = kk * 32;
                v16h a0 = frag_a_f(ar0, k0 + kb);
                v16h a1 = frag_a_f(ar1, k0 + kb);
                const _Float16* wrow = Wl + (size_t)(k0 + kb) * GATES + g0 + lrow;
                v16h b0, b1;
#pragma unroll
                for (int i = 0; i < 8; ++i) {
                    b0[i] = wrow[(size_t)i * GATES]; b0[i + 8] = wrow[(size_t)(16 + i) * GATES];
                    b1[i] = wrow[(size_t)i * GATES + 16]; b1[i + 8] = wrow[(size_t)(16 + i) * GATES + 16];
                }
                __builtin_prefetch(wrow + (size_t)32 * GATES, 0, 3);
                acc00 = __builtin_amdgcn_wmma_f32_16x16x32_f16(false, a0, false, b0, (short)0, acc00, false, false);
                acc01 = __builtin_amdgcn_wmma_f32_16x16x32_f16(false, a0, false, b1, (short)0, acc01, false, false);
                acc10 = __builtin_amdgcn_wmma_f32_16x16x32_f16(false, a1, false, b0, (short)0, acc10, false, false);
                acc11 = __builtin_amdgcn_wmma_f32_16x16x32_f16(false, a1, false, b1, (short)0, acc11, false, false);
            }
        } else {
            const _Float16* ar0 = curR + (size_t)(m0 + lrow) * HID;
            const _Float16* ar1 = curR + (size_t)(m0 + 16 + lrow) * HID;
#pragma unroll 4
            for (int kk = 0; kk < 8; ++kk) {
                const int k0 = kk * 32;
                v16h a0 = frag_a_h(ar0, k0 + kb);
                v16h a1 = frag_a_h(ar1, k0 + kb);
                const _Float16* wrow = Wl + (size_t)(k0 + kb) * GATES + g0 + lrow;
                v16h b0, b1;
#pragma unroll
                for (int i = 0; i < 8; ++i) {
                    b0[i] = wrow[(size_t)i * GATES]; b0[i + 8] = wrow[(size_t)(16 + i) * GATES];
                    b1[i] = wrow[(size_t)i * GATES + 16]; b1[i + 8] = wrow[(size_t)(16 + i) * GATES + 16];
                }
                __builtin_prefetch(wrow + (size_t)32 * GATES, 0, 3);
                acc00 = __builtin_amdgcn_wmma_f32_16x16x32_f16(false, a0, false, b0, (short)0, acc00, false, false);
                acc01 = __builtin_amdgcn_wmma_f32_16x16x32_f16(false, a0, false, b1, (short)0, acc01, false, false);
                acc10 = __builtin_amdgcn_wmma_f32_16x16x32_f16(false, a1, false, b0, (short)0, acc10, false, false);
                acc11 = __builtin_amdgcn_wmma_f32_16x16x32_f16(false, a1, false, b1, (short)0, acc11, false, false);
            }
        }

        // ---- recurrent part of K (k = 256..511) ----
        {
            const _Float16* ar0 = scRj + (size_t)(m0 + lrow) * HID;
            const _Float16* ar1 = scRj + (size_t)(m0 + 16 + lrow) * HID;
#pragma unroll 4
            for (int kk = 0; kk < 8; ++kk) {
                const int k0 = kk * 32;
                v16h a0 = frag_a_h(ar0, k0 + kb);
                v16h a1 = frag_a_h(ar1, k0 + kb);
                const _Float16* wrow = Wl + (size_t)(HID + k0 + kb) * GATES + g0 + lrow;
                v16h b0, b1;
#pragma unroll
                for (int i = 0; i < 8; ++i) {
                    b0[i] = wrow[(size_t)i * GATES]; b0[i + 8] = wrow[(size_t)(16 + i) * GATES];
                    b1[i] = wrow[(size_t)i * GATES + 16]; b1[i + 8] = wrow[(size_t)(16 + i) * GATES + 16];
                }
                __builtin_prefetch(wrow + (size_t)32 * GATES, 0, 3);
                acc00 = __builtin_amdgcn_wmma_f32_16x16x32_f16(false, a0, false, b0, (short)0, acc00, false, false);
                acc01 = __builtin_amdgcn_wmma_f32_16x16x32_f16(false, a0, false, b1, (short)0, acc01, false, false);
                acc10 = __builtin_amdgcn_wmma_f32_16x16x32_f16(false, a1, false, b0, (short)0, acc10, false, false);
                acc11 = __builtin_amdgcn_wmma_f32_16x16x32_f16(false, a1, false, b1, (short)0, acc11, false, false);
            }
        }

        // Spill gate tiles to LDS with bias (C/D layout: M = p + 8*lhalf, N = lrow)
        const float bias0 = bc[(size_t)j * GATES + g0 + lrow];
        const float bias1 = bc[(size_t)j * GATES + g0 + 16 + lrow];
#pragma unroll
        for (int p = 0; p < 8; ++p) {
            const int r0 = p + 8 * lhalf;
            smem[r0     ][wv * 32 + lrow     ] = acc00[p] + bias0;
            smem[r0     ][wv * 32 + 16 + lrow] = acc01[p] + bias1;
            smem[r0 + 16][wv * 32 + lrow     ] = acc10[p] + bias0;
            smem[r0 + 16][wv * 32 + 16 + lrow] = acc11[p] + bias1;
        }
        __syncthreads();

        // Elementwise, faithful to the reference's h/c swap:
        //   c_ret = sig(f)*sh + sig(i)*tanh(g); h_ret = sig(o)*tanh(c_ret)
        //   sh[j] <- c_ret (fp32), cur <- c_ret (f16), sc[j] <- h_ret (f16)
        const int row = tid >> 2;         // 0..31
        const int c0  = (tid & 3) * 8;    // 0,8,16,24
#pragma unroll
        for (int q = 0; q < 8; ++q) {
            const int cl = c0 + q;
            float ii = smem[row][cl];
            float ff = smem[row][32 + cl];
            float gg = smem[row][64 + cl];
            float oo = smem[row][96 + cl];
            const size_t gidx = ((size_t)j * BATCH + (m0 + row)) * HID + (h0 + cl);
            float c_old = sh32[gidx];
            float c_new = sigm(ff) * c_old + sigm(ii) * tanhf(gg);
            float h_new = sigm(oo) * tanhf(c_new);
            sh32[gidx] = c_new;
            scW[gidx]  = (_Float16)h_new;
            curW[(size_t)(m0 + row) * HID + (h0 + cl)] = (_Float16)c_new;
        }

        if (j + 1 < NL) {
            // Device-wide barrier before next layer reads curW/scW
            __threadfence();
            __syncthreads();
            if (tid == 0) {
                atomicAdd(&ctr[j], 1u);
                while (__hip_atomic_load(&ctr[j], __ATOMIC_ACQUIRE,
                                         __HIP_MEMORY_SCOPE_AGENT) < gridDim.x) { }
            }
            __syncthreads();
            __threadfence();
        }
    }
}

extern "C" void kernel_launch(void* const* d_in, const int* in_sizes, int n_in,
                              void* d_out, int out_size, void* d_ws, size_t ws_size,
                              hipStream_t stream) {
    (void)in_sizes; (void)n_in; (void)out_size; (void)ws_size;

    const float* x   = (const float*)d_in[0];
    const float* Wih = (const float*)d_in[1];
    const float* Whh = (const float*)d_in[2];
    const float* bih = (const float*)d_in[3];
    const float* bhh = (const float*)d_in[4];

    char* ws = (char*)d_ws;
    size_t off = 0;
    _Float16* Wc   = (_Float16*)(ws + off); off += (size_t)NL * KTOT * GATES * sizeof(_Float16);     // 10.0 MB
    float*    bc   = (float*)   (ws + off); off += (size_t)NL * GATES * sizeof(float);               // 40 KB
    _Float16* sc16 = (_Float16*)(ws + off); off += (size_t)2 * NL * BATCH * HID * sizeof(_Float16);  // 2.6 MB ping-pong
    _Float16* cur0 = (_Float16*)(ws + off); off += (size_t)BATCH * HID * sizeof(_Float16);
    _Float16* cur1 = (_Float16*)(ws + off); off += (size_t)BATCH * HID * sizeof(_Float16);
    float*    sh32 = (float*)   (ws + off); off += (size_t)NL * BATCH * HID * sizeof(float);         // 2.6 MB
    unsigned* ctr  = (unsigned*)(ws + off); off += (size_t)SEQ * NL * sizeof(unsigned);              // 40 KB

    // Re-initialize every call (harness poisons ws once; graph replays must be deterministic)
    hipMemsetAsync(sc16, 0, (size_t)2 * NL * BATCH * HID * sizeof(_Float16), stream);
    hipMemsetAsync(sh32, 0, (size_t)NL * BATCH * HID * sizeof(float), stream);
    hipMemsetAsync(ctr,  0, (size_t)SEQ * NL * sizeof(unsigned), stream);

    { size_t N = (size_t)NL * KTOT * GATES;
      pack_w<<<(unsigned)((N + 255) / 256), 256, 0, stream>>>(Wih, Whh, Wc); }
    pack_b<<<(NL * GATES + 255) / 256, 256, 0, stream>>>(bih, bhh, bc);

    for (int t = 0; t < SEQ; ++t) {
        const _Float16* scRd = sc16 + (size_t)(t & 1)       * NL * BATCH * HID;
        _Float16*       scWr = sc16 + (size_t)((t + 1) & 1) * NL * BATCH * HID;
        lstm_step<<<64, 128, 0, stream>>>(x, t, Wc, bc, scRd, scWr, cur0, cur1, sh32,
                                          ctr + (size_t)t * NL);
    }

    // Output = sh[L-1] (fp32, 256x256)
    hipMemcpyAsync(d_out, sh32 + (size_t)(NL - 1) * BATCH * HID,
                   (size_t)BATCH * HID * sizeof(float),
                   hipMemcpyDeviceToDevice, stream);
}
